// Transformer_33122787787524
// MI455X (gfx1250) — matrix operand order repaired and checked by reading, not compiled, not run on hardware
//
#include <hip/hip_runtime.h>

// ---------------------------------------------------------------------------
// CDNA5 (gfx1250) transformer forward pass.
// - bf16 WMMA (v_wmma_f32_16x16x32_bf16) for all GEMMs, f32 accumulate
// - Tensor Data Mover (tensor_load_to_lds) for layout-preserving tile loads
// - transposed LDS staging so every fragment load is ds_load_b128
// ---------------------------------------------------------------------------

typedef __attribute__((ext_vector_type(16))) __bf16 bf16x16;
typedef __attribute__((ext_vector_type(8)))  float  f32x8;
typedef __attribute__((ext_vector_type(4)))  unsigned int u32x4;
typedef __attribute__((ext_vector_type(8)))  int   i32x8;
typedef __attribute__((ext_vector_type(4)))  int   i32x4;

union FragA { bf16x16 v; unsigned short u[16]; };

static __device__ __forceinline__ unsigned short f2bf(float f) {
  unsigned int u = __float_as_uint(f);
  u = (u + 0x7FFFu + ((u >> 16) & 1u)) >> 16;   // round-to-nearest-even
  return (unsigned short)u;
}
static __device__ __forceinline__ float bf2f(unsigned short h) {
  return __uint_as_float(((unsigned int)h) << 16);
}

// ---------------------------------------------------------------------------
// TDM: load a (rows x 32) bf16 tile (row stride = stride_elems in global) into
// LDS at lds_byte_addr, padded to 40 elems (80B) per row via D# padding
// (pad_interval = 16 DWORDs -> pad after each 64B row, pad_amount = 4 DWORDs).
// One wave issues this; completion via S_WAIT_TENSORCNT.
// ---------------------------------------------------------------------------
static __device__ __forceinline__ void tdm_load_tile32(
    const void* gsrc, unsigned lds_byte_addr, unsigned rows, unsigned stride_elems)
{
  unsigned long long ga = (unsigned long long)(uintptr_t)gsrc;
  u32x4 g0;
  g0.x = 1u;                                                  // count=1, user D#
  g0.y = lds_byte_addr;                                       // LDS byte address
  g0.z = (unsigned)ga;                                        // global addr lo
  g0.w = ((unsigned)(ga >> 32) & 0x01FFFFFFu) | (2u << 30);   // addr hi | type=2
  i32x8 g1;
  g1[0] = (int)((1u << 16)        // data_size = 2 bytes
              | (1u << 20)        // pad_enable
              | (3u << 22)        // pad_interval: 16 DWORDs (64B rows)
              | (3u << 25));      // pad_amount: 4 DWORDs (16B)
  g1[1] = (int)((stride_elems & 0xFFFFu) << 16);                    // tensor_dim0 lo
  g1[2] = (int)((stride_elems >> 16) | ((rows & 0xFFFFu) << 16));   // dim0 hi|dim1 lo
  g1[3] = (int)((rows >> 16) | (32u << 16));                        // dim1 hi|tile_dim0=32
  g1[4] = (int)(rows & 0xFFFFu);                                    // tile_dim1|tile_dim2=0
  g1[5] = (int)stride_elems;                                        // dim0_stride lo32
  g1[6] = 0;                                                        // stride hi|d1stride lo
  g1[7] = 0;
  i32x4 gz = {0, 0, 0, 0};
#if defined(__clang_major__) && (__clang_major__ >= 23)
  i32x8 gz8 = {0, 0, 0, 0, 0, 0, 0, 0};
  __builtin_amdgcn_tensor_load_to_lds(g0, g1, gz, gz, gz8, 0);
#else
  __builtin_amdgcn_tensor_load_to_lds(g0, g1, gz, gz, 0);
#endif
}

// ---------------------------------------------------------------------------
// Generic GEMM: Out[b][m][n] = act( sum_k W[m][k]*X[b][k][n] + bias[m] + res )
// W: MxK bf16 row-major (shared across batch). X: batch x K x N bf16.
// Block 128(M) x 64(N), K-step 32; 8 waves, each 32x32 via 2x2 WMMA.
// A tile via TDM (wave 0). B tile staged transposed so B-frags are ds_load_b128.
// flags: bit0 = ReLU, bit1 = transposed f32 store (out[b][n][m]).
// ---------------------------------------------------------------------------
__global__ __launch_bounds__(256) void gemm_bf16_kernel(
    const unsigned short* __restrict__ W,
    const unsigned short* __restrict__ X,
    const float* __restrict__ bias,
    const unsigned short* __restrict__ res,
    unsigned short* __restrict__ outB,
    float* __restrict__ outF,
    int M, int N, int K, int flags)
{
  __shared__ unsigned short As[128][40];  // [m][k]  (TDM-filled, padded rows)
  __shared__ unsigned short Bt[64][40];   // [n][k]  (transposed staging)

  const int tid  = threadIdx.x;
  const int lane = tid & 31;
  const int wave = tid >> 5;
  const int wm   = wave & 3;      // 4 M-groups of 32
  const int wn   = wave >> 2;     // 2 N-groups of 32
  const int b    = blockIdx.z;
  const int m0   = blockIdx.y * 128;
  const int n0   = blockIdx.x * 64;
  const unsigned short* Xb = X + (size_t)b * K * N;

  const int am = lane & 15, akb = (lane >> 4) * 16;   // A-frag mapping
  const int bn = lane & 15, bkb = (lane >> 4) * 16;   // B-frag mapping

  f32x8 zero = {0.f,0.f,0.f,0.f,0.f,0.f,0.f,0.f};
  f32x8 acc[2][2];
  acc[0][0]=zero; acc[0][1]=zero; acc[1][0]=zero; acc[1][1]=zero;

  for (int k0 = 0; k0 < K; k0 += 32) {
    if (wave == 0) {   // A tile 128x32 via Tensor Data Mover
      tdm_load_tile32(W + (size_t)m0 * K + k0,
                      (unsigned)(uintptr_t)&As[0][0], 128u, (unsigned)K);
    }
    // B tile 32x64: coalesced global load, transposed LDS store
    {
      int r = tid >> 3, cb = (tid & 7) * 8;
      uint4 qv = *(const uint4*)&Xb[(size_t)(k0 + r) * N + n0 + cb];
      const unsigned short* e = (const unsigned short*)&qv;
      #pragma unroll
      for (int j = 0; j < 8; ++j) Bt[cb + j][r] = e[j];
    }
    if (k0 + 32 < K)   // gfx1250 global_prefetch_b8 for next B tile
      __builtin_prefetch(&Xb[(size_t)(k0 + 32 + (tid >> 3)) * N + n0], 0, 0);
    if (wave == 0) __builtin_amdgcn_s_wait_tensorcnt(0);
    __syncthreads();

    FragA a[2], bb[2];
    #pragma unroll
    for (int f = 0; f < 2; ++f) {
      int row = wm * 32 + f * 16 + am;
      #pragma unroll
      for (int i = 0; i < 8; ++i) {
        a[f].u[i]     = As[row][akb + i];
        a[f].u[8 + i] = As[row][akb + 8 + i];
      }
      int col = wn * 32 + f * 16 + bn;
      #pragma unroll
      for (int i = 0; i < 16; ++i) bb[f].u[i] = Bt[col][bkb + i];
    }
    #pragma unroll
    for (int fm = 0; fm < 2; ++fm)
      #pragma unroll
      for (int fn = 0; fn < 2; ++fn)
        acc[fm][fn] = __builtin_amdgcn_wmma_f32_16x16x32_bf16(
            false, a[fm].v, false, bb[fn].v, (short)0, acc[fm][fn], false, false);
    __syncthreads();
  }

  const int cn = lane & 15, cmb = (lane >> 4) * 8;
  #pragma unroll
  for (int fm = 0; fm < 2; ++fm)
    #pragma unroll
    for (int fn = 0; fn < 2; ++fn) {
      int gcol = n0 + wn * 32 + fn * 16 + cn;
      #pragma unroll
      for (int r = 0; r < 8; ++r) {
        int grow = m0 + wm * 32 + fm * 16 + cmb + r;
        float v = acc[fm][fn][r];
        if (bias) v += bias[grow];
        size_t oi = (size_t)b * M * N + (size_t)grow * N + gcol;
        if (res) v += bf2f(res[oi]);
        if (flags & 1) v = v > 0.f ? v : 0.f;
        if (outB) outB[oi] = f2bf(v);
        if (outF) {
          if (flags & 2) outF[(size_t)b * M * N + (size_t)gcol * M + grow] = v;
          else           outF[oi] = v;
        }
      }
    }
}

// ---------------------------------------------------------------------------
// Attention scores: S[bh][q][k] = 0.125 * sum_d Q[b][h*64+d][q] * K[b][h*64+d][k]
// Q,K staged transposed into LDS ([q][d], [k][d]) -> contiguous frag loads.
// mode: 0 = pad(q)&pad(k); 1 = + causal (k<=q); 2 = cross.
// ---------------------------------------------------------------------------
__global__ __launch_bounds__(256) void attn_scores_kernel(
    const unsigned short* __restrict__ Q,    // (B, U, Lq)
    const unsigned short* __restrict__ Km,   // (B, U, Lk)
    const int* __restrict__ padq,            // (B, Lq)
    const int* __restrict__ padk,            // (B, Lk)
    float* __restrict__ S,                   // (B*H, Lq, Lk)
    int Lq, int Lk, int U, int mode)
{
  __shared__ unsigned short Qt[128][40];  // [q][d]
  __shared__ unsigned short Kt[64][40];   // [k][d]

  const int tid = threadIdx.x, lane = tid & 31, wave = tid >> 5;
  const int wm = wave & 3, wn = wave >> 2;
  const int bh = blockIdx.z, b = bh >> 3, h = bh & 7;
  const int q0 = blockIdx.y * 128, k0 = blockIdx.x * 64;
  const unsigned short* Qb = Q + ((size_t)b * U + h * 64) * Lq;
  const unsigned short* Kb = Km + ((size_t)b * U + h * 64) * Lk;

  const int am = lane & 15, akb = (lane >> 4) * 16;
  const int bn = lane & 15, bkb = (lane >> 4) * 16;

  f32x8 zero = {0.f,0.f,0.f,0.f,0.f,0.f,0.f,0.f};
  f32x8 acc[2][2];
  acc[0][0]=zero; acc[0][1]=zero; acc[1][0]=zero; acc[1][1]=zero;

  for (int d0 = 0; d0 < 64; d0 += 32) {
    #pragma unroll
    for (int it = tid; it < 512; it += 256) {      // Qt: 32(d) x 128(q) transposed
      int r = it >> 4, cb = (it & 15) * 8;
      uint4 qv = *(const uint4*)&Qb[(size_t)(d0 + r) * Lq + q0 + cb];
      const unsigned short* e = (const unsigned short*)&qv;
      #pragma unroll
      for (int j = 0; j < 8; ++j) Qt[cb + j][r] = e[j];
    }
    {                                              // Kt: 32(d) x 64(k) transposed
      int r = tid >> 3, cb = (tid & 7) * 8;
      uint4 qv = *(const uint4*)&Kb[(size_t)(d0 + r) * Lk + k0 + cb];
      const unsigned short* e = (const unsigned short*)&qv;
      #pragma unroll
      for (int j = 0; j < 8; ++j) Kt[cb + j][r] = e[j];
    }
    __syncthreads();

    FragA a[2], bb[2];
    #pragma unroll
    for (int f = 0; f < 2; ++f) {
      int qq = wm * 32 + f * 16 + am;              // A[m=q][k=d] = Qt[q][d]
      #pragma unroll
      for (int i = 0; i < 8; ++i) {
        a[f].u[i]     = Qt[qq][akb + i];
        a[f].u[8 + i] = Qt[qq][akb + 8 + i];
      }
      int kk = wn * 32 + f * 16 + bn;              // B[k=d][n=k] = Kt[k][d]
      #pragma unroll
      for (int i = 0; i < 16; ++i) bb[f].u[i] = Kt[kk][bkb + i];
    }
    #pragma unroll
    for (int fm = 0; fm < 2; ++fm)
      #pragma unroll
      for (int fn = 0; fn < 2; ++fn)
        acc[fm][fn] = __builtin_amdgcn_wmma_f32_16x16x32_bf16(
            false, a[fm].v, false, bb[fn].v, (short)0, acc[fm][fn], false, false);
    __syncthreads();
  }

  const int cn = lane & 15, cmb = (lane >> 4) * 8;
  #pragma unroll
  for (int fm = 0; fm < 2; ++fm)
    #pragma unroll
    for (int fn = 0; fn < 2; ++fn) {
      int gk = k0 + wn * 32 + fn * 16 + cn;
      #pragma unroll
      for (int r = 0; r < 8; ++r) {
        int gq = q0 + wm * 32 + fm * 16 + cmb + r;
        float v = acc[fm][fn][r] * 0.125f;         // dk=64 -> 1/sqrt(64)
        bool ok = (padq[b * Lq + gq] != 0) && (padk[b * Lk + gk] != 0);
        if (mode == 1) ok = ok && (gk <= gq);      // causal
        S[((size_t)bh * Lq + gq) * Lk + gk] = ok ? v : -1e30f;
      }
    }
}

// ---------------------------------------------------------------------------
// Row softmax (wave per row, Lk==256 -> 8 elems/lane), output bf16 probs.
// ---------------------------------------------------------------------------
__global__ __launch_bounds__(256) void softmax_kernel(
    const float* __restrict__ S, unsigned short* __restrict__ P, int Lk)
{
  const int wave = threadIdx.x >> 5, lane = threadIdx.x & 31;
  const size_t row = (size_t)blockIdx.x * 8 + wave;
  const float* s = S + row * Lk;
  unsigned short* p = P + row * Lk;

  float vals[8];
  float mx = -3.0e38f;
  #pragma unroll
  for (int i = 0; i < 8; ++i) { vals[i] = s[lane + i * 32]; mx = fmaxf(mx, vals[i]); }
  #pragma unroll
  for (int off = 16; off; off >>= 1) mx = fmaxf(mx, __shfl_xor(mx, off, 32));
  float sum = 0.f;
  #pragma unroll
  for (int i = 0; i < 8; ++i) { float e = __expf(vals[i] - mx); vals[i] = e; sum += e; }
  #pragma unroll
  for (int off = 16; off; off >>= 1) sum += __shfl_xor(sum, off, 32);
  float inv = (mx < -1e29f) ? 0.f : (1.f / sum);
  #pragma unroll
  for (int i = 0; i < 8; ++i) p[lane + i * 32] = f2bf(vals[i] * inv);
}

// ---------------------------------------------------------------------------
// A*V: C[b][h*64+d][q] = sum_k P[bh][q][k] * V[b][h*64+d][k]
// Both tiles are layout-preserving copies -> both loaded via TDM.
// M=64(d), N tile 128(q), reduction Lk in steps of 32. 8 waves: 2(M) x 4(N).
// ---------------------------------------------------------------------------
__global__ __launch_bounds__(256) void attn_av_kernel(
    const unsigned short* __restrict__ P,   // (B*H, Lq, Lk) bf16 probs
    const unsigned short* __restrict__ V,   // (B, U, Lk)
    unsigned short* __restrict__ C,         // (B, U, Lq)
    int Lq, int Lk, int U)
{
  __shared__ unsigned short Vs[64][40];     // [d][k]
  __shared__ unsigned short Ps[128][40];    // [q][k]

  const int tid = threadIdx.x, lane = tid & 31, wave = tid >> 5;
  const int wm = wave & 1, wn = wave >> 1;
  const int bh = blockIdx.z, b = bh >> 3, h = bh & 7;
  const int n0 = blockIdx.x * 128;
  const unsigned short* Vb = V + ((size_t)b * U + h * 64) * Lk;
  const unsigned short* Pb = P + (size_t)bh * Lq * Lk;
  unsigned short* Cb = C + ((size_t)b * U + h * 64) * Lq;

  const int am = lane & 15, akb = (lane >> 4) * 16;
  const int bn = lane & 15, bkb = (lane >> 4) * 16;

  f32x8 zero = {0.f,0.f,0.f,0.f,0.f,0.f,0.f,0.f};
  f32x8 acc[2][2];
  acc[0][0]=zero; acc[0][1]=zero; acc[1][0]=zero; acc[1][1]=zero;

  for (int k0 = 0; k0 < Lk; k0 += 32) {
    if (wave == 0) {        // P tile 128x32 via TDM
      tdm_load_tile32(Pb + (size_t)n0 * Lk + k0,
                      (unsigned)(uintptr_t)&Ps[0][0], 128u, (unsigned)Lk);
      __builtin_amdgcn_s_wait_tensorcnt(0);
    } else if (wave == 1) { // V tile 64x32 via TDM
      tdm_load_tile32(Vb + k0,
                      (unsigned)(uintptr_t)&Vs[0][0], 64u, (unsigned)Lk);
      __builtin_amdgcn_s_wait_tensorcnt(0);
    }
    __syncthreads();

    FragA a[2], bb[2];
    #pragma unroll
    for (int f = 0; f < 2; ++f) {
      int dd = wm * 32 + f * 16 + am;              // A[m=d][k] = Vs[d][k]
      #pragma unroll
      for (int i = 0; i < 8; ++i) {
        a[f].u[i]     = Vs[dd][akb + i];
        a[f].u[8 + i] = Vs[dd][akb + 8 + i];
      }
      int qq = wn * 32 + f * 16 + bn;              // B[k][n=q] = Ps[q][k]
      #pragma unroll
      for (int i = 0; i < 16; ++i) bb[f].u[i] = Ps[qq][bkb + i];
    }
    #pragma unroll
    for (int fm = 0; fm < 2; ++fm)
      #pragma unroll
      for (int fn = 0; fn < 2; ++fn)
        acc[fm][fn] = __builtin_amdgcn_wmma_f32_16x16x32_bf16(
            false, a[fm].v, false, bb[fn].v, (short)0, acc[fm][fn], false, false);
    __syncthreads();
  }

  const int cn = lane & 15, cmb = (lane >> 4) * 8;
  #pragma unroll
  for (int fm = 0; fm < 2; ++fm)
    #pragma unroll
    for (int fn = 0; fn < 2; ++fn) {
      int gq = n0 + wn * 32 + fn * 16 + cn;
      #pragma unroll
      for (int r = 0; r < 8; ++r) {
        int gd = wm * 32 + fm * 16 + cmb + r;
        Cb[(size_t)gd * Lq + gq] = f2bf(acc[fm][fn][r]);
      }
    }
}

// ---------------------------------------------------------------------------
// Small VALU kernels
// ---------------------------------------------------------------------------
__global__ void embed_kernel(const int* __restrict__ tok,
                             const float* __restrict__ emb,
                             const float* __restrict__ pos,   // (U, MAXLEN)
                             unsigned short* __restrict__ e,  // (B, U, L)
                             int B, int L, int U, int maxlen, float scale)
{
  int idx = blockIdx.x * blockDim.x + threadIdx.x;
  if (idx >= B * U * L) return;
  int l = idx % L;
  int u = (idx / L) % U;
  int b = idx / (L * U);
  int t = tok[b * L + l];
  float v = emb[(size_t)t * U + u] * scale + pos[(size_t)u * maxlen + l];
  e[idx] = f2bf(v);
}

__global__ void pad_kernel(const int* __restrict__ tok, int* __restrict__ flag, int n)
{
  int i = blockIdx.x * blockDim.x + threadIdx.x;
  if (i < n) flag[i] = (tok[i] != 0) ? 1 : 0;
}

__global__ void cast_kernel(const float* __restrict__ src,
                            unsigned short* __restrict__ dst, size_t n)
{
  size_t i = (size_t)blockIdx.x * blockDim.x + threadIdx.x;
  size_t stride = (size_t)gridDim.x * blockDim.x;
  for (; i < n; i += stride) dst[i] = f2bf(src[i]);
}

// ---------------------------------------------------------------------------
// Host orchestration
// ---------------------------------------------------------------------------
extern "C" void kernel_launch(void* const* d_in, const int* in_sizes, int n_in,
                              void* d_out, int out_size, void* d_ws, size_t ws_size,
                              hipStream_t stream) {
  (void)in_sizes; (void)n_in; (void)out_size; (void)ws_size;
  const int B = 8, L = 256, U = 512, H = 8, F = 2048, NL = 6, V = 32000, MAXLEN = 500;

  const int*   x      = (const int*)  d_in[0];
  const int*   y      = (const int*)  d_in[1];
  const float* embX   = (const float*)d_in[2];
  const float* embY   = (const float*)d_in[3];
  const float* posEnc = (const float*)d_in[4];
  const float* encAttnF = (const float*)d_in[5];
  const float* encW1F   = (const float*)d_in[6];
  const float* encB1    = (const float*)d_in[7];
  const float* encW2F   = (const float*)d_in[8];
  const float* encB2    = (const float*)d_in[9];
  const float* decSelfF = (const float*)d_in[10];
  const float* decSrcF  = (const float*)d_in[11];
  const float* decW1F   = (const float*)d_in[12];
  const float* decB1    = (const float*)d_in[13];
  const float* decW2F   = (const float*)d_in[14];
  const float* decB2    = (const float*)d_in[15];
  const float* affF     = (const float*)d_in[16];

  // workspace carve-out (256B aligned)
  size_t off = 0;
  char* base = (char*)d_ws;
  auto take = [&](size_t elems, size_t esz) -> void* {
    void* p = base + off;
    off += ((elems * esz + 255) & ~(size_t)255);
    return p;
  };
  unsigned short* wEncAttn = (unsigned short*)take((size_t)NL * 4 * U * U, 2);
  unsigned short* wEncW1   = (unsigned short*)take((size_t)NL * F * U, 2);
  unsigned short* wEncW2   = (unsigned short*)take((size_t)NL * U * F, 2);
  unsigned short* wDecSelf = (unsigned short*)take((size_t)NL * 4 * U * U, 2);
  unsigned short* wDecSrc  = (unsigned short*)take((size_t)NL * 4 * U * U, 2);
  unsigned short* wDecW1   = (unsigned short*)take((size_t)NL * F * U, 2);
  unsigned short* wDecW2   = (unsigned short*)take((size_t)NL * U * F, 2);
  unsigned short* wAff     = (unsigned short*)take((size_t)V * U, 2);
  unsigned short* eEnc = (unsigned short*)take((size_t)B * U * L, 2);
  unsigned short* eDec = (unsigned short*)take((size_t)B * U * L, 2);
  unsigned short* qbuf = (unsigned short*)take((size_t)B * U * L, 2);
  unsigned short* kbuf = (unsigned short*)take((size_t)B * U * L, 2);
  unsigned short* vbuf = (unsigned short*)take((size_t)B * U * L, 2);
  unsigned short* cbuf = (unsigned short*)take((size_t)B * U * L, 2);
  unsigned short* hid  = (unsigned short*)take((size_t)B * F * L, 2);
  float*          Sbuf = (float*)         take((size_t)B * H * L * L, 4);
  unsigned short* Pbuf = (unsigned short*)take((size_t)B * H * L * L, 2);
  int* px = (int*)take((size_t)B * L, 4);
  int* py = (int*)take((size_t)B * L, 4);

  auto cast = [&](const float* src, unsigned short* dst, size_t n) {
    int blocks = (int)((n + 255) / 256); if (blocks > 8192) blocks = 8192;
    cast_kernel<<<blocks, 256, 0, stream>>>(src, dst, n);
  };
  cast(encAttnF, wEncAttn, (size_t)NL * 4 * U * U);
  cast(encW1F,   wEncW1,   (size_t)NL * F * U);
  cast(encW2F,   wEncW2,   (size_t)NL * U * F);
  cast(decSelfF, wDecSelf, (size_t)NL * 4 * U * U);
  cast(decSrcF,  wDecSrc,  (size_t)NL * 4 * U * U);
  cast(decW1F,   wDecW1,   (size_t)NL * F * U);
  cast(decW2F,   wDecW2,   (size_t)NL * U * F);
  cast(affF,     wAff,     (size_t)V * U);

  pad_kernel<<<(B * L + 255) / 256, 256, 0, stream>>>(x, px, B * L);
  pad_kernel<<<(B * L + 255) / 256, 256, 0, stream>>>(y, py, B * L);

  const float scaleEmb = 22.62741699796952f;  // sqrt(512)
  embed_kernel<<<(B * U * L + 255) / 256, 256, 0, stream>>>(x, embX, posEnc, eEnc, B, L, U, MAXLEN, scaleEmb);
  embed_kernel<<<(B * U * L + 255) / 256, 256, 0, stream>>>(y, embY, posEnc, eDec, B, L, U, MAXLEN, scaleEmb);

  auto gemm = [&](const unsigned short* Wm, const unsigned short* Xm, const float* bias,
                  const unsigned short* res, unsigned short* outB, float* outF,
                  int M, int N, int K, int flags) {
    dim3 g(N / 64, M / 128, B);
    gemm_bf16_kernel<<<g, 256, 0, stream>>>(Wm, Xm, bias, res, outB, outF, M, N, K, flags);
  };

  auto mha = [&](const unsigned short* W4, unsigned short* xq, const unsigned short* z,
                 const int* pq, const int* pk, int mode) {
    gemm(W4 + 0 * (size_t)U * U, xq, nullptr, nullptr, qbuf, nullptr, U, L, U, 0);
    gemm(W4 + 1 * (size_t)U * U, z,  nullptr, nullptr, kbuf, nullptr, U, L, U, 0);
    gemm(W4 + 2 * (size_t)U * U, z,  nullptr, nullptr, vbuf, nullptr, U, L, U, 0);
    attn_scores_kernel<<<dim3(L / 64, L / 128, B * H), 256, 0, stream>>>(
        qbuf, kbuf, pq, pk, Sbuf, L, L, U, mode);
    softmax_kernel<<<(B * H * L) / 8, 256, 0, stream>>>(Sbuf, Pbuf, L);
    attn_av_kernel<<<dim3(L / 128, 1, B * H), 256, 0, stream>>>(Pbuf, vbuf, cbuf, L, L, U);
    gemm(W4 + 3 * (size_t)U * U, cbuf, nullptr, xq, xq, nullptr, U, L, U, 0);
  };

  auto ffn = [&](const unsigned short* W1, const float* b1,
                 const unsigned short* W2, const float* b2, unsigned short* e) {
    gemm(W1, e,   b1, nullptr, hid, nullptr, F, L, U, 1);   // ReLU
    gemm(W2, hid, b2, e,       e,   nullptr, U, L, F, 0);   // +residual
  };

  // encoder
  for (int i = 0; i < NL; ++i) {
    mha(wEncAttn + (size_t)i * 4 * U * U, eEnc, eEnc, px, px, 0);
    ffn(wEncW1 + (size_t)i * F * U, encB1 + (size_t)i * F,
        wEncW2 + (size_t)i * U * F, encB2 + (size_t)i * U, eEnc);
  }
  // decoder
  for (int i = 0; i < NL; ++i) {
    mha(wDecSelf + (size_t)i * 4 * U * U, eDec, eDec, py, py, 1);  // causal self
    mha(wDecSrc  + (size_t)i * 4 * U * U, eDec, eEnc, py, px, 2);  // cross
    ffn(wDecW1 + (size_t)i * F * U, decB1 + (size_t)i * F,
        wDecW2 + (size_t)i * U * F, decB2 + (size_t)i * U, eDec);
  }

  // logits[b][l][v] = sum_u d[b][u][l] * Waff[v][u]  (transposed f32 store)
  gemm(wAff, eDec, nullptr, nullptr, nullptr, (float*)d_out, V, L, U, 2);
}
